// WeightedVectorization_17652315586731
// MI455X (gfx1250) — matrix operand order repaired and checked by reading, not compiled, not run on hardware
//
#include <hip/hip_runtime.h>
#include <stdint.h>

typedef __attribute__((ext_vector_type(16))) __bf16 v16bf;
typedef __attribute__((ext_vector_type(8)))  float  v8f;

#define Bn 256
#define Dn 256
#define On 1024
#define PITCH 264   // LDS row pitch in ushorts: 528B = 132 banks -> lane stride 4 banks, conflict-free b128

__device__ __forceinline__ uint32_t f2bf1(float f) {
  // round-to-nearest-even f32 -> bf16 bits
  uint32_t u = __float_as_uint(f);
  return (u + 0x7FFFu + ((u >> 16) & 1u)) >> 16;
}
__device__ __forceinline__ uint32_t pack2(float lo, float hi) {
  return f2bf1(lo) | (f2bf1(hi) << 16);
}

// Pre-pass: Kt[o*Dn + j] = bf16(K[j*On + o])  (transpose + convert, so that
// a K-matrix column (the WMMA B operand / weighting tile) is contiguous)
__global__ void k_transpose_bf16(const float* __restrict__ K,
                                 uint16_t* __restrict__ Kt) {
  int idx = blockIdx.x * 256 + threadIdx.x;   // 0 .. Dn*On-1
  int o = idx >> 8;                           // 0..1023
  int j = idx & 255;                          // 0..255  (fastest -> coalesced writes)
  Kt[o * Dn + j] = (uint16_t)f2bf1(K[j * On + o]);
}

union FragU {
  v16bf v;
  uint4 q[2];
};

__global__ __launch_bounds__(256, 2)
void k_quadform(const float* __restrict__ A,      // (B, D, D) f32
                const uint16_t* __restrict__ Kt,  // (O, D) bf16 bits
                float* __restrict__ out) {        // (B, O) f32
  __shared__ uint16_t lds[Dn * PITCH];            // 132 KB padded bf16 copy of A_b

  const int b = blockIdx.x;

  // ---- Stage A_b -> LDS as bf16 (coalesced b128 global loads, conflict-free b64 ds stores)
  {
    const float4* src = (const float4*)(A + (size_t)b * Dn * Dn);
    #pragma unroll 4
    for (int q = 0; q < 64; ++q) {
      int c = q * 256 + threadIdx.x;            // 16384 float4 chunks
      int row = c >> 6;                         // 64 chunks per 256-wide row
      int col = c & 63;
      float4 v = src[c];
      *(uint2*)(lds + row * PITCH + col * 4) =
          make_uint2(pack2(v.x, v.y), pack2(v.z, v.w));
    }
  }
  __syncthreads();

  const int lane = threadIdx.x & 31;
  const int wid  = threadIdx.x >> 5;            // 8 waves
  const int lo   = lane & 15;                   // row (A) / column (B,C) selector
  const int hi   = lane >> 4;                   // lane-half selector

  #pragma unroll 1
  for (int ot = 0; ot < 4; ++ot) {
    // Two o-tiles per pass: each A fragment feeds two WMMAs (halves LDS traffic per FLOP).
    const int o_tile0 = wid * 8 + ot * 2;
    const int o_tile1 = o_tile0 + 1;
    const uint16_t* KtRow0 = Kt + (size_t)(o_tile0 * 16 + lo) * Dn;
    const uint16_t* KtRow1 = Kt + (size_t)(o_tile1 * 16 + lo) * Dn;

    // B fragments for the whole K-dimension, both o-tiles, kept in VGPRs (128 VGPRs).
    // 32x16 bf16 B layout: lanes 0-15 hold K=0..15 (contiguous), lanes 16-31 K=16..31.
    FragU B0[8], B1[8];
    #pragma unroll
    for (int jf = 0; jf < 8; ++jf) {
      const uint4* p0 = (const uint4*)(KtRow0 + jf * 32 + hi * 16);
      const uint4* p1 = (const uint4*)(KtRow1 + jf * 32 + hi * 16);
      B0[jf].q[0] = p0[0];
      B0[jf].q[1] = p0[1];
      B1[jf].q[0] = p1[0];
      B1[jf].q[1] = p1[1];
    }

    v8f acc0 = {};                              // weighted accumulators, C layout
    v8f acc1 = {};
    #pragma unroll 1
    for (int it = 0; it < 16; ++it) {
      v8f c0 = {};
      v8f c1 = {};
      const uint16_t* Arow = lds + (it * 16 + lo) * PITCH;
      #pragma unroll
      for (int jf = 0; jf < 8; ++jf) {
        // 16x32 bf16 A layout: lanes 0-15 K=0..15, lanes 16-31 K=16..31 (contiguous, matches B)
        FragU Af;
        Af.q[0] = *(const uint4*)(Arow + jf * 32 + hi * 16);
        Af.q[1] = *(const uint4*)(Arow + jf * 32 + hi * 16 + 8);
        c0 = __builtin_amdgcn_wmma_f32_16x16x32_bf16(false, Af.v, false, B0[jf].v,
                                                     (short)0, c0, false, false);
        c1 = __builtin_amdgcn_wmma_f32_16x16x32_bf16(false, Af.v, false, B1[jf].v,
                                                     (short)0, c1, false, false);
      }
      // Weighted accumulate: acc[m,n] += K[i_base+m, o_base+n] * tmp[m,n].
      // C layout: lane -> N=lo, M = r + 8*hi  => one contiguous b128 from each Kt row.
      uint4 wq0 = *(const uint4*)(KtRow0 + it * 16 + hi * 8);
      uint4 wq1 = *(const uint4*)(KtRow1 + it * 16 + hi * 8);
      uint32_t wb0[4] = {wq0.x, wq0.y, wq0.z, wq0.w};
      uint32_t wb1[4] = {wq1.x, wq1.y, wq1.z, wq1.w};
      #pragma unroll
      for (int k = 0; k < 4; ++k) {
        acc0[2 * k]     += __uint_as_float(wb0[k] << 16)         * c0[2 * k];
        acc0[2 * k + 1] += __uint_as_float(wb0[k] & 0xFFFF0000u) * c0[2 * k + 1];
        acc1[2 * k]     += __uint_as_float(wb1[k] << 16)         * c1[2 * k];
        acc1[2 * k + 1] += __uint_as_float(wb1[k] & 0xFFFF0000u) * c1[2 * k + 1];
      }
    }

    // Reduce over M: lanes<16 hold M=0..7, lanes>=16 hold M=8..15 (same N). xor-16 swizzle combines.
    float s0 = acc0[0] + acc0[1] + acc0[2] + acc0[3] + acc0[4] + acc0[5] + acc0[6] + acc0[7];
    float s1 = acc1[0] + acc1[1] + acc1[2] + acc1[3] + acc1[4] + acc1[5] + acc1[6] + acc1[7];
    s0 += __int_as_float(__builtin_amdgcn_ds_swizzle(__float_as_int(s0), 0x401F));
    s1 += __int_as_float(__builtin_amdgcn_ds_swizzle(__float_as_int(s1), 0x401F));
    if (hi == 0) {
      out[(size_t)b * On + o_tile0 * 16 + lo] = s0;
      out[(size_t)b * On + o_tile1 * 16 + lo] = s1;
    }
  }
}

extern "C" void kernel_launch(void* const* d_in, const int* in_sizes, int n_in,
                              void* d_out, int out_size, void* d_ws, size_t ws_size,
                              hipStream_t stream) {
  const float* inputs = (const float*)d_in[0];   // (256, 256, 256) f32
  const float* kernel = (const float*)d_in[1];   // (256, 1024) f32
  float* out = (float*)d_out;                    // (256, 1024) f32
  uint16_t* Kt = (uint16_t*)d_ws;                // (1024, 256) bf16 bits, 512 KB

  k_transpose_bf16<<<(Dn * On) / 256, 256, 0, stream>>>(kernel, Kt);
  k_quadform<<<Bn, 256, 0, stream>>>(inputs, Kt, out);
}
